// SwinBlock_53919019434529
// MI455X (gfx1250) — matrix operand order repaired and checked by reading, not compiled, not run on hardware
//
#include <hip/hip_runtime.h>
#include <math.h>

// ---------------------------------------------------------------------------
// Swin block for MI455X (gfx1250, wave32). All GEMMs use
// v_wmma_f32_16x16x32_f16 (f16 inputs, f32 accumulate).
// ---------------------------------------------------------------------------

typedef _Float16 h16;
typedef __attribute__((ext_vector_type(8)))  _Float16 v8h;
typedef __attribute__((ext_vector_type(16))) _Float16 v16h;
typedef __attribute__((ext_vector_type(8)))  float    v8f;

union V16H { v16h v; v8h h[2]; _Float16 e[16]; };
union V8F  { v8f  v; float f[8]; };

#define CDIM   192
#define HIDDEN 384
#define NHEAD  6
#define HD     32
#define NSIDE  28          // windows per image side (224/8)
#define WPI    784         // windows per image
#define NWIN   6272        // total windows (8*784)
#define NTOK   401408      // total tokens
#define IMG    224

// LDS strides chosen for bank-conflict-free access (see analysis)
#define RAWLD  193         // f32 row stride (193 % 64 == 1)
#define ALD    200         // f16 row stride (stride bytes % 256 = 144 -> conflict free frag loads)

// ---------------- scratch layout (bytes, 256-aligned) ----------------------
static constexpr size_t OFF_WQ   = 0;                       // 576*192 f16
static constexpr size_t OFF_WP   = 221184;                  // 192*192 f16
static constexpr size_t OFF_W1   = 294912;                  // 384*192 f16
static constexpr size_t OFF_W2   = 442368;                  // 192*384 f16
static constexpr size_t OFF_BIAS = 589824;                  // 6*64*64 f32
static constexpr size_t OFF_Q    = 688128;                  // NTOK*192 f16 (per-head layout)
static constexpr size_t OFF_K    = OFF_Q  + 154140672;
static constexpr size_t OFF_VT   = OFF_K  + 154140672;
static constexpr size_t OFF_O    = OFF_VT + 154140672;
static constexpr size_t OFF_H    = OFF_O  + 154140672;      // NTOK*192 f32
static constexpr size_t OFF_A1   = OFF_Q;                   // alias: NTOK*384 f16 over dead Q+K

// ---------------- fragment helpers (CDNA5 WMMA layouts) --------------------
// A (16-bit, 16x32): lane m=l&15; halves: K in [h*8, h*8+8) and [16+h*8, ...)
__device__ __forceinline__ v16h ldA(const h16* __restrict__ p, int ld, int lane) {
  const int m = lane & 15, hf = lane >> 4;
  V16H u;
  const h16* r = p + (size_t)m * ld + hf * 8;
  u.h[0] = *(const v8h*)(r);
  u.h[1] = *(const v8h*)(r + 16);
  return u.v;
}
// B (16-bit, 32x16): lane n=l&15; halves: K in [h*8, h*8+8) and [16+h*8, ...) of column n.
// Stored as row-major W[n][k]  (i.e. B = W^T), so per-lane two 16B chunks.
__device__ __forceinline__ v16h ldB(const h16* __restrict__ p, int ld, int lane) {
  const int n = lane & 15, hf = lane >> 4;
  V16H u;
  const h16* r = p + (size_t)n * ld + hf * 8;
  u.h[0] = *(const v8h*)(r);
  u.h[1] = *(const v8h*)(r + 16);
  return u.v;
}
__device__ __forceinline__ v8f wmma16(v16h a, v16h b, v8f c) {
  return __builtin_amdgcn_wmma_f32_16x16x32_f16(false, a, false, b, (short)0, c,
                                                false, false);
}
__device__ __forceinline__ v8f zero8() { v8f z = {0.f,0.f,0.f,0.f,0.f,0.f,0.f,0.f}; return z; }

// token -> source/dest pixel (roll by +4 with wrap; identical for gather & scatter)
__device__ __forceinline__ void tok2pix(int gt, int& b, int& hh, int& ww) {
  int g = gt >> 6, t = gt & 63;
  b = g / WPI; int r = g - b * WPI;
  int wy = r / NSIDE, wx = r - wy * NSIDE;
  int hp = wy * 8 + (t >> 3) + 4; if (hp >= IMG) hp -= IMG;
  int wp = wx * 8 + (t & 7) + 4;  if (wp >= IMG) wp -= IMG;
  hh = hp; ww = wp;
}

// ---------------------------------------------------------------------------
// prep: weights f32->f16, bias table [head][qi][kj]
// ---------------------------------------------------------------------------
__global__ void k_prep(const float* __restrict__ qkv_w, const float* __restrict__ proj_w,
                       const float* __restrict__ w1,    const float* __restrict__ w2,
                       const float* __restrict__ rel_table, const int* __restrict__ rel_index,
                       h16* wq, h16* wp, h16* w1h, h16* w2h, float* biasT) {
  const int i = blockIdx.x * blockDim.x + threadIdx.x;
  const int st = gridDim.x * blockDim.x;
  for (int k = i; k < 576*192; k += st) wq[k]  = (h16)qkv_w[k];
  for (int k = i; k < 192*192; k += st) wp[k]  = (h16)proj_w[k];
  for (int k = i; k < 384*192; k += st) w1h[k] = (h16)w1[k];
  for (int k = i; k < 192*384; k += st) w2h[k] = (h16)w2[k];
  for (int k = i; k < NHEAD*64*64; k += st) {
    int hd = k >> 12, rem = k & 4095;
    biasT[k] = rel_table[rel_index[rem] * NHEAD + hd];
  }
}

// ---------------------------------------------------------------------------
// Kernel 1: gather + LN1 + QKV GEMM (M=256, N=576, K=192)
// ---------------------------------------------------------------------------
__global__ __launch_bounds__(256) void k_qkv(const float* __restrict__ x,
    const float* __restrict__ g1, const float* __restrict__ b1,
    const float* __restrict__ qkv_b, const h16* __restrict__ wq,
    h16* __restrict__ Q, h16* __restrict__ K, h16* __restrict__ Vt) {
  extern __shared__ char smem[];
  float* raw = (float*)smem;                                  // 256*RAWLD f32
  h16*   A   = (h16*)(smem + 256*RAWLD*4);                    // 256*ALD f16
  float* sg  = (float*)(smem + 256*RAWLD*4 + 256*ALD*2);
  float* sb  = sg + CDIM;
  const int tid = threadIdx.x;
  if (tid < CDIM) { sg[tid] = g1[tid]; sb[tid] = b1[tid]; }

  const int gt = blockIdx.x * 256 + tid;
  int b, hh, ww; tok2pix(gt, b, hh, ww);
  const float* xp = x + ((size_t)b * CDIM) * IMG * IMG + (size_t)hh * IMG + ww;
  float s1 = 0.f, s2 = 0.f;
  for (int c = 0; c < CDIM; ++c) {
    float v = xp[(size_t)c * IMG * IMG];
    raw[tid * RAWLD + c] = v; s1 += v; s2 += v * v;
  }
  const float mean = s1 * (1.f / CDIM);
  const float inv  = rsqrtf(s2 * (1.f / CDIM) - mean * mean + 1e-5f);
  __syncthreads();
  for (int c = 0; c < CDIM; ++c)
    A[tid * ALD + c] = (h16)((raw[tid * RAWLD + c] - mean) * inv * sg[c] + sb[c]);
  __syncthreads();

  const int wv = tid >> 5, lane = tid & 31;
  const int mtb = (wv & 3) * 4, ntb = (wv >> 2) * 18;
  const float scale = 0.17677669529663687f;   // hd^-0.5
  for (int nti = 0; nti < 18; ++nti) {
    const int nt = ntb + nti;
    V8F acc[4];
#pragma unroll
    for (int i = 0; i < 4; ++i) acc[i].v = zero8();
    for (int kt = 0; kt < 6; ++kt) {
      v16h bf = ldB(wq + (size_t)(nt * 16) * CDIM + kt * 32, CDIM, lane);
#pragma unroll
      for (int mi = 0; mi < 4; ++mi) {
        v16h af = ldA(A + (size_t)((mtb + mi) * 16) * ALD + kt * 32, ALD, lane);
        acc[mi].v = wmma16(af, bf, acc[mi].v);
      }
    }
    const int n = nt * 16 + (lane & 15);
    const float bias = qkv_b[n];
    const int region = n / CDIM, cc = n - region * CDIM;
    const int head = cc >> 5, d = cc & 31;
#pragma unroll
    for (int mi = 0; mi < 4; ++mi)
#pragma unroll
      for (int r = 0; r < 8; ++r) {
        const int m = (mtb + mi) * 16 + r + ((lane >> 4) << 3);
        const int g = blockIdx.x * 256 + m;
        const int win = g >> 6, t = g & 63;
        const float v = acc[mi].f[r] + bias;
        const size_t qoff = (((size_t)win * NHEAD + head) * 64 + t) * HD + d;
        if (region == 0)      Q[qoff] = (h16)(v * scale);
        else if (region == 1) K[qoff] = (h16)v;
        else Vt[(((size_t)win * NHEAD + head) * HD + d) * 64 + t] = (h16)v;
      }
  }
}

// ---------------------------------------------------------------------------
// Kernel 2: attention per window; 8 waves x 3 (head, m-block) tasks
// ---------------------------------------------------------------------------
__global__ __launch_bounds__(256) void k_attn(const h16* __restrict__ Q,
    const h16* __restrict__ K, const h16* __restrict__ Vt,
    const float* __restrict__ biasT, h16* __restrict__ O) {
  extern __shared__ char smem[];
  const int tid = threadIdx.x, wv = tid >> 5, lane = tid & 31;
  h16* P = (h16*)smem + wv * 16 * 72;          // per-wave 16x72 f16
  const int win = blockIdx.x;

  for (int task = wv; task < 24; task += 8) {
    const int head = task >> 2, mtb = task & 3;
    const h16* Qb = Q  + (((size_t)win * NHEAD + head) * 64 + mtb * 16) * HD;
    const h16* Kb = K  +  ((size_t)win * NHEAD + head) * 64 * HD;
    const h16* Vb = Vt +  ((size_t)win * NHEAD + head) * HD * 64;

    const v16h aq = ldA(Qb, HD, lane);          // K-dim = hd = 32: one frag
    V8F s[4];
#pragma unroll
    for (int nt = 0; nt < 4; ++nt) {
      v16h kb = ldB(Kb + (size_t)(nt * 16) * HD, HD, lane);
      s[nt].v = wmma16(aq, kb, zero8());
    }
    const float* bp = biasT + (size_t)head * 4096;
#pragma unroll
    for (int r = 0; r < 8; ++r) {
      const int qi = mtb * 16 + r + ((lane >> 4) << 3);
      float vals[4], mx = -1e30f;
#pragma unroll
      for (int nt = 0; nt < 4; ++nt) {
        const int kj = nt * 16 + (lane & 15);
        float v = s[nt].f[r] + bp[(size_t)qi * 64 + kj];
        vals[nt] = v; mx = fmaxf(mx, v);
      }
      for (int off = 8; off >= 1; off >>= 1) mx = fmaxf(mx, __shfl_xor(mx, off, 32));
      float sum = 0.f;
#pragma unroll
      for (int nt = 0; nt < 4; ++nt) { vals[nt] = __expf(vals[nt] - mx); sum += vals[nt]; }
      for (int off = 8; off >= 1; off >>= 1) sum += __shfl_xor(sum, off, 32);
      const float rs = 1.f / sum;
      const int lr = r + ((lane >> 4) << 3);
#pragma unroll
      for (int nt = 0; nt < 4; ++nt)
        P[lr * 72 + nt * 16 + (lane & 15)] = (h16)(vals[nt] * rs);
    }
    asm volatile("s_wait_dscnt 0" ::: "memory");   // P store -> A-frag load (same wave)
#pragma unroll
    for (int nt = 0; nt < 2; ++nt) {
      V8F o; o.v = zero8();
#pragma unroll
      for (int kt = 0; kt < 2; ++kt) {
        v16h ap = ldA(P + kt * 32, 72, lane);
        v16h vb = ldB(Vb + (size_t)(nt * 16) * 64 + kt * 32, 64, lane);
        o.v = wmma16(ap, vb, o.v);
      }
      const int d = nt * 16 + (lane & 15);
#pragma unroll
      for (int r = 0; r < 8; ++r) {
        const int t = mtb * 16 + r + ((lane >> 4) << 3);
        O[((size_t)win * 64 + t) * CDIM + head * HD + d] = (h16)o.f[r];
      }
    }
  }
}

// ---------------------------------------------------------------------------
// Kernel 3: proj GEMM (M=256,N=192,K=192) + residual (gathered x) -> h
// ---------------------------------------------------------------------------
__global__ __launch_bounds__(256) void k_proj(const float* __restrict__ x,
    const h16* __restrict__ O, const h16* __restrict__ wp,
    const float* __restrict__ pb, float* __restrict__ Hs) {
  const int tid = threadIdx.x, wv = tid >> 5, lane = tid & 31;
  const int mtb = (wv & 3) * 4, ntb = (wv >> 2) * 6;
  const size_t rowbase = (size_t)blockIdx.x * 256;
  for (int nti = 0; nti < 6; ++nti) {
    const int nt = ntb + nti;
    V8F acc[4];
#pragma unroll
    for (int i = 0; i < 4; ++i) acc[i].v = zero8();
    for (int kt = 0; kt < 6; ++kt) {
      v16h bf = ldB(wp + (size_t)(nt * 16) * CDIM + kt * 32, CDIM, lane);
#pragma unroll
      for (int mi = 0; mi < 4; ++mi) {
        v16h af = ldA(O + (rowbase + (mtb + mi) * 16) * CDIM + kt * 32, CDIM, lane);
        acc[mi].v = wmma16(af, bf, acc[mi].v);
      }
    }
    const int n = nt * 16 + (lane & 15);
    const float bias = pb[n];
#pragma unroll
    for (int mi = 0; mi < 4; ++mi)
#pragma unroll
      for (int r = 0; r < 8; ++r) {
        const int g = (int)rowbase + (mtb + mi) * 16 + r + ((lane >> 4) << 3);
        int b, hh, ww; tok2pix(g, b, hh, ww);
        const float v = acc[mi].f[r] + bias +
            x[(((size_t)b * CDIM + n) * IMG + hh) * IMG + ww];
        Hs[(size_t)g * CDIM + n] = v;
      }
  }
}

// ---------------------------------------------------------------------------
// Kernel 4: LN2 + FC1 GEMM (M=256,N=384,K=192) + GELU
// ---------------------------------------------------------------------------
__global__ __launch_bounds__(256) void k_fc1(const float* __restrict__ Hs,
    const float* __restrict__ g2, const float* __restrict__ b2,
    const float* __restrict__ bb1, const h16* __restrict__ w1h,
    h16* __restrict__ a1) {
  extern __shared__ char smem[];
  float* raw = (float*)smem;
  h16*   A   = (h16*)(smem + 256*RAWLD*4);
  float* sg  = (float*)(smem + 256*RAWLD*4 + 256*ALD*2);
  float* sb  = sg + CDIM;
  const int tid = threadIdx.x;
  if (tid < CDIM) { sg[tid] = g2[tid]; sb[tid] = b2[tid]; }
  const int gt = blockIdx.x * 256 + tid;
  float s1 = 0.f, s2 = 0.f;
  for (int c = 0; c < CDIM; ++c) {
    float v = Hs[(size_t)gt * CDIM + c];
    raw[tid * RAWLD + c] = v; s1 += v; s2 += v * v;
  }
  const float mean = s1 * (1.f / CDIM);
  const float inv  = rsqrtf(s2 * (1.f / CDIM) - mean * mean + 1e-5f);
  __syncthreads();
  for (int c = 0; c < CDIM; ++c)
    A[tid * ALD + c] = (h16)((raw[tid * RAWLD + c] - mean) * inv * sg[c] + sb[c]);
  __syncthreads();

  const int wv = tid >> 5, lane = tid & 31;
  const int mtb = (wv & 3) * 4, ntb = (wv >> 2) * 12;
  for (int nti = 0; nti < 12; ++nti) {
    const int nt = ntb + nti;
    V8F acc[4];
#pragma unroll
    for (int i = 0; i < 4; ++i) acc[i].v = zero8();
    for (int kt = 0; kt < 6; ++kt) {
      v16h bf = ldB(w1h + (size_t)(nt * 16) * CDIM + kt * 32, CDIM, lane);
#pragma unroll
      for (int mi = 0; mi < 4; ++mi) {
        v16h af = ldA(A + (size_t)((mtb + mi) * 16) * ALD + kt * 32, ALD, lane);
        acc[mi].v = wmma16(af, bf, acc[mi].v);
      }
    }
    const int n = nt * 16 + (lane & 15);
    const float bias = bb1[n];
#pragma unroll
    for (int mi = 0; mi < 4; ++mi)
#pragma unroll
      for (int r = 0; r < 8; ++r) {
        const int g = blockIdx.x * 256 + (mtb + mi) * 16 + r + ((lane >> 4) << 3);
        float v = acc[mi].f[r] + bias;
        v = 0.5f * v * (1.f + erff(v * 0.70710678118654752f));  // exact GELU
        a1[(size_t)g * HIDDEN + n] = (h16)v;
      }
  }
}

// ---------------------------------------------------------------------------
// Kernel 5: FC2 GEMM (M=256,N=192,K=384) + residuals + scatter -> out (NCHW)
// ---------------------------------------------------------------------------
__global__ __launch_bounds__(256) void k_fc2(const float* __restrict__ x,
    const float* __restrict__ Hs, const h16* __restrict__ a1,
    const h16* __restrict__ w2h, const float* __restrict__ bb2,
    float* __restrict__ out) {
  const int tid = threadIdx.x, wv = tid >> 5, lane = tid & 31;
  const int mtb = (wv & 3) * 4, ntb = (wv >> 2) * 6;
  const size_t rowbase = (size_t)blockIdx.x * 256;
  for (int nti = 0; nti < 6; ++nti) {
    const int nt = ntb + nti;
    V8F acc[4];
#pragma unroll
    for (int i = 0; i < 4; ++i) acc[i].v = zero8();
    for (int kt = 0; kt < 12; ++kt) {
      v16h bf = ldB(w2h + (size_t)(nt * 16) * HIDDEN + kt * 32, HIDDEN, lane);
#pragma unroll
      for (int mi = 0; mi < 4; ++mi) {
        v16h af = ldA(a1 + (rowbase + (mtb + mi) * 16) * HIDDEN + kt * 32, HIDDEN, lane);
        acc[mi].v = wmma16(af, bf, acc[mi].v);
      }
    }
    const int n = nt * 16 + (lane & 15);
    const float bias = bb2[n];
#pragma unroll
    for (int mi = 0; mi < 4; ++mi)
#pragma unroll
      for (int r = 0; r < 8; ++r) {
        const int g = (int)rowbase + (mtb + mi) * 16 + r + ((lane >> 4) << 3);
        int b, hh, ww; tok2pix(g, b, hh, ww);
        const size_t pix = (((size_t)b * CDIM + n) * IMG + hh) * IMG + ww;
        out[pix] = x[pix] + Hs[(size_t)g * CDIM + n] + acc[mi].f[r] + bias;
      }
  }
}

// ---------------------------------------------------------------------------
extern "C" void kernel_launch(void* const* d_in, const int* in_sizes, int n_in,
                              void* d_out, int out_size, void* d_ws, size_t ws_size,
                              hipStream_t stream) {
  (void)in_sizes; (void)n_in; (void)out_size; (void)ws_size;
  const float* x      = (const float*)d_in[0];
  const float* qkv_w  = (const float*)d_in[1];
  const float* qkv_b  = (const float*)d_in[2];
  const float* proj_w = (const float*)d_in[3];
  const float* proj_b = (const float*)d_in[4];
  const float* rel_t  = (const float*)d_in[5];
  const float* g1     = (const float*)d_in[6];
  const float* b1     = (const float*)d_in[7];
  const float* g2     = (const float*)d_in[8];
  const float* b2     = (const float*)d_in[9];
  const float* w1     = (const float*)d_in[10];
  const float* bb1    = (const float*)d_in[11];
  const float* w2     = (const float*)d_in[12];
  const float* bb2    = (const float*)d_in[13];
  const int*   relidx = (const int*)d_in[14];
  float* out = (float*)d_out;

  char* ws = (char*)d_ws;
  h16*   wq    = (h16*)(ws + OFF_WQ);
  h16*   wp    = (h16*)(ws + OFF_WP);
  h16*   w1h   = (h16*)(ws + OFF_W1);
  h16*   w2h   = (h16*)(ws + OFF_W2);
  float* biasT = (float*)(ws + OFF_BIAS);
  h16*   Q     = (h16*)(ws + OFF_Q);
  h16*   K     = (h16*)(ws + OFF_K);
  h16*   Vt    = (h16*)(ws + OFF_VT);
  h16*   O     = (h16*)(ws + OFF_O);
  float* Hs    = (float*)(ws + OFF_H);
  h16*   a1    = (h16*)(ws + OFF_A1);      // aliases dead Q+K

  const size_t smemLN = 256*RAWLD*4 + 256*ALD*2 + 2*CDIM*4;   // ~301.5 KB (<320KB WGP LDS)
  const size_t smemAT = 8 * 16 * 72 * 2;

  k_prep<<<512, 256, 0, stream>>>(qkv_w, proj_w, w1, w2, rel_t, relidx,
                                  wq, wp, w1h, w2h, biasT);
  k_qkv <<<NWIN/4, 256, smemLN, stream>>>(x, g1, b1, qkv_b, wq, Q, K, Vt);
  k_attn<<<NWIN,   256, smemAT, stream>>>(Q, K, Vt, biasT, O);
  k_proj<<<NWIN/4, 256, 0,      stream>>>(x, O, wp, proj_b, Hs);
  k_fc1 <<<NWIN/4, 256, smemLN, stream>>>(Hs, g2, b2, bb1, w1h, a1);
  k_fc2 <<<NWIN/4, 256, 0,      stream>>>(x, Hs, a1, w2h, bb2, out);
}
